// MultiScaleAttention_36292473651403
// MI455X (gfx1250) — matrix operand order repaired and checked by reading, not compiled, not run on hardware
//
#include <hip/hip_runtime.h>

typedef __attribute__((ext_vector_type(16))) __bf16 v16bf;
typedef __attribute__((ext_vector_type(8)))  __bf16 v8bf;
typedef __attribute__((ext_vector_type(8)))  float  v8f;
typedef __attribute__((ext_vector_type(4)))  float  v4f;

#define BB   4
#define NH   6
#define HD   32
#define DIMM 192
#define HH   56
#define WW   56
#define NQ   3136
#define HPK  28
#define NK   784
#define BH   24                    // B * NUM_HEADS
#define SCALE 0.17677669529663687f // 32^-0.5

// ---------- wave-level reductions (16-lane halves, wave32) ----------
__device__ __forceinline__ float rmax16(float v) {
#pragma unroll
    for (int m = 1; m < 16; m <<= 1) v = fmaxf(v, __shfl_xor(v, m, 32));
    return v;
}
__device__ __forceinline__ float rsum16(float v) {
#pragma unroll
    for (int m = 1; m < 16; m <<= 1) v += __shfl_xor(v, m, 32);
    return v;
}

// ---------- weight prep: transpose + f32->bf16 ----------
__global__ void prep_weights(const float* __restrict__ wqkv, const float* __restrict__ wproj,
                             __bf16* __restrict__ wqkv_t, __bf16* __restrict__ wproj_t) {
    int i = blockIdx.x * blockDim.x + threadIdx.x;
    if (i < 576 * DIMM) {
        int n = i / DIMM, k = i % DIMM;
        wqkv_t[i] = (__bf16)wqkv[(size_t)k * 576 + n];
    }
    int j = i - 576 * DIMM;
    if (j >= 0 && j < DIMM * DIMM) {
        int n = j / DIMM, k = j % DIMM;
        wproj_t[j] = (__bf16)wproj[(size_t)k * DIMM + n];
    }
}

// ---------- QKV GEMM: [12544,192] x [192,576], wave = 16x32 tile ----------
__global__ __launch_bounds__(256)
void qkv_gemm(const float* __restrict__ x, const __bf16* __restrict__ wt,
              const float* __restrict__ bias, float* __restrict__ qkv /*[3][B][6][NQ][32]*/) {
    const int lane = threadIdx.x & 31;
    const int ln   = lane & 15;
    const int half = lane >> 4;
    const int gw   = (blockIdx.x * blockDim.x + threadIdx.x) >> 5;
    const int mt   = gw / 18;   // 784 m-tiles of 16
    const int nt   = gw % 18;   // 18 n-tiles of 32
    if (mt >= 784) return;
    const int row = mt * 16 + ln;
    const int n0  = nt * 32;
    const int ca  = half ? 16 : 0;
    const int cb  = half ? 16 : 0;
    v8f acc0 = {}, acc1 = {};
#pragma unroll
    for (int kk = 0; kk < DIMM; kk += 32) {
        const float* ap = x + (size_t)row * DIMM + kk + ca;
        v4f x0 = *(const v4f*)(ap);
        v4f x1 = *(const v4f*)(ap + 4);
        v4f x2 = *(const v4f*)(ap + 8);
        v4f x3 = *(const v4f*)(ap + 12);
        v16bf a;
#pragma unroll
        for (int i = 0; i < 4; i++) {
            a[i]      = (__bf16)x0[i];
            a[4 + i]  = (__bf16)x1[i];
            a[8 + i]  = (__bf16)x2[i];
            a[12 + i] = (__bf16)x3[i];
        }
        v16bf b0 = *(const v16bf*)(wt + (size_t)(n0 + ln) * DIMM + kk + cb);
        v16bf b1 = *(const v16bf*)(wt + (size_t)(n0 + 16 + ln) * DIMM + kk + cb);
        acc0 = __builtin_amdgcn_wmma_f32_16x16x32_bf16(false, a, false, b0, (short)0, acc0, false, false);
        acc1 = __builtin_amdgcn_wmma_f32_16x16x32_bf16(false, a, false, b1, (short)0, acc1, false, false);
    }
#pragma unroll
    for (int sub = 0; sub < 2; sub++) {
        v8f& acc = sub ? acc1 : acc0;
        int col = n0 + sub * 16 + ln;
        int s = col / DIMM, rm = col % DIMM, h = rm / HD, d = rm % HD;
        float bv = bias[col];
#pragma unroll
        for (int r = 0; r < 8; r++) {
            int m = mt * 16 + r + half * 8;
            int bidx = m / NQ, n = m % NQ;
            qkv[((((size_t)s * BB + bidx) * NH + h) * NQ + n) * HD + d] = acc[r] + bv;
        }
    }
}

// ---------- depthwise 3x3 pool + LayerNorm: wave per pixel, lane per channel ----------
__global__ __launch_bounds__(256)
void pool_ln(const float* __restrict__ src, const float* __restrict__ wconv,
             const float* __restrict__ gamma, const float* __restrict__ beta,
             int stride, int Hp, int Wp,
             __bf16* __restrict__ out_bf, int transposed, float* __restrict__ out_f32) {
    const int lane = threadIdx.x & 31;  // channel d
    const int gw   = (blockIdx.x * blockDim.x + threadIdx.x) >> 5;
    const int Np   = Hp * Wp;
    if (gw >= BH * Np) return;
    const int bh = gw / Np;
    const int p  = gw % Np;
    const int py = p / Wp, px = p % Wp;
    float acc = 0.f;
#pragma unroll
    for (int ky = 0; ky < 3; ky++) {
        int iy = py * stride + ky - 1;
        if (iy < 0 || iy >= HH) continue;
#pragma unroll
        for (int kx = 0; kx < 3; kx++) {
            int ix = px * stride + kx - 1;
            if (ix < 0 || ix >= WW) continue;
            acc += src[((size_t)bh * NQ + iy * WW + ix) * HD + lane] * wconv[(ky * 3 + kx) * HD + lane];
        }
    }
    // LayerNorm across 32 channels = full wave
    float mu = acc;
#pragma unroll
    for (int m = 1; m < 32; m <<= 1) mu += __shfl_xor(mu, m, 32);
    mu *= (1.0f / 32.0f);
    float dv = acc - mu;
    float var = dv * dv;
#pragma unroll
    for (int m = 1; m < 32; m <<= 1) var += __shfl_xor(var, m, 32);
    var *= (1.0f / 32.0f);
    float y = dv * rsqrtf(var + 1e-5f) * gamma[lane] + beta[lane];
    if (transposed) out_bf[((size_t)bh * HD + lane) * Np + p] = (__bf16)y;   // V^T for PV B-operand
    else            out_bf[((size_t)bh * Np + p) * HD + lane] = (__bf16)y;
    if (out_f32)    out_f32[((size_t)bh * Np + p) * HD + lane] = y;
}

// ---------- flash attention: wave = 16 q-rows of one (b,h) ----------
__global__ __launch_bounds__(256)
void attention(const __bf16* __restrict__ qb, const float* __restrict__ qf,
               const __bf16* __restrict__ kb, const __bf16* __restrict__ vt,
               __bf16* __restrict__ attn_out /*[B][NQ][192]*/) {
    __shared__ __bf16 lds_p[8][16 * 32];   // per-wave P staging tile
    const int lane = threadIdx.x & 31;
    const int ln   = lane & 15;
    const int half = lane >> 4;
    const int wid  = threadIdx.x >> 5;
    const int gw   = blockIdx.x * 8 + wid;
    const int qt   = gw % 196;
    const int bh   = gw / 196;
    __bf16* lp = &lds_p[wid][0];

    // A-operand: Q tile 16x32 (row-major bf16)
    const int ca = half ? 16 : 0;
    const __bf16* qp = qb + ((size_t)bh * NQ + qt * 16 + ln) * HD + ca;
    v8bf qlo = *(const v8bf*)qp;
    v8bf qhi = *(const v8bf*)(qp + 8);
    v16bf a_q;
#pragma unroll
    for (int i = 0; i < 8; i++) { a_q[i] = qlo[i]; a_q[8 + i] = qhi[i]; }

    v8f o0 = {}, o1 = {};
    float m_i[8], l_i[8];
#pragma unroll
    for (int r = 0; r < 8; r++) { m_i[r] = -3.0e38f; l_i[r] = 0.f; }

    for (int it = 0; it < 25; it++) {   // 25 * 32 covers NK=784 with masking
        const int kt = it * 32;
        // ---- S = Q K^T (two 16-column tiles); B-operand = rows of K
        int kn0 = kt + ln;
        int kn1 = kt + 16 + ln;
        const __bf16* kp0 = kb + ((size_t)bh * NK + (kn0 < NK ? kn0 : NK - 1)) * HD + (half ? 16 : 0);
        const __bf16* kp1 = kb + ((size_t)bh * NK + (kn1 < NK ? kn1 : NK - 1)) * HD + (half ? 16 : 0);
        v16bf bk0 = *(const v16bf*)kp0;
        v16bf bk1 = *(const v16bf*)kp1;
        v8f s0 = {}, s1 = {};
        s0 = __builtin_amdgcn_wmma_f32_16x16x32_bf16(false, a_q, false, bk0, (short)0, s0, false, false);
        s1 = __builtin_amdgcn_wmma_f32_16x16x32_bf16(false, a_q, false, bk1, (short)0, s1, false, false);
        const float NEG = -3.0e38f;
        float t0[8], t1[8];
#pragma unroll
        for (int r = 0; r < 8; r++) {
            t0[r] = (kn0 < NK) ? s0[r] * SCALE : NEG;
            t1[r] = (kn1 < NK) ? s1[r] * SCALE : NEG;
        }
        // ---- online softmax (row r lives in lanes 0-15 / r+8 in lanes 16-31)
#pragma unroll
        for (int r = 0; r < 8; r++) {
            float rm = rmax16(fmaxf(t0[r], t1[r]));
            float mn = fmaxf(m_i[r], rm);
            float f  = __expf(m_i[r] - mn);
            m_i[r] = mn;
            l_i[r] *= f;
            o0[r] *= f;
            o1[r] *= f;
            float p0 = __expf(t0[r] - mn);
            float p1 = __expf(t1[r] - mn);
            l_i[r] += rsum16(p0 + p1);
            int rowm = r + half * 8;
            lp[rowm * 32 + ln]      = (__bf16)p0;   // C-layout -> row-major LDS
            lp[rowm * 32 + 16 + ln] = (__bf16)p1;
        }
        asm volatile("s_wait_dscnt 0x0" ::: "memory");   // intra-wave cross-lane LDS
        // ---- reload P as A-operand (16x32)
        const __bf16* pp = lp + ln * 32 + ca;
        v8bf plo = *(const v8bf*)pp;
        v8bf phi = *(const v8bf*)(pp + 8);
        v16bf a_p;
#pragma unroll
        for (int i = 0; i < 8; i++) { a_p[i] = plo[i]; a_p[8 + i] = phi[i]; }
        // ---- O += P V : B-operand = rows of V^T (contiguous kn runs)
        int knb = kt + (half ? 16 : 0);
        if (knb > NK - 16) knb = NK - 16;   // clamped; masked P cols are 0
        const __bf16* vp0 = vt + ((size_t)bh * HD + ln) * NK + knb;
        const __bf16* vp1 = vt + ((size_t)bh * HD + 16 + ln) * NK + knb;
        v16bf bv0 = *(const v16bf*)vp0;
        v16bf bv1 = *(const v16bf*)vp1;
        o0 = __builtin_amdgcn_wmma_f32_16x16x32_bf16(false, a_p, false, bv0, (short)0, o0, false, false);
        o1 = __builtin_amdgcn_wmma_f32_16x16x32_bf16(false, a_p, false, bv1, (short)0, o1, false, false);
        asm volatile("s_wait_dscnt 0x0" ::: "memory");   // keep next stores behind reads
    }
    // ---- epilogue: normalize, residual add, write [B][n][h*32+d] bf16
    const int bidx = bh / NH, h = bh % NH;
#pragma unroll
    for (int r = 0; r < 8; r++) {
        int qrow = qt * 16 + r + half * 8;
        float inv = 1.0f / l_i[r];
        const float* qfp = qf + ((size_t)bh * NQ + qrow) * HD;
        float v0 = o0[r] * inv + qfp[ln];
        float v1 = o1[r] * inv + qfp[16 + ln];
        size_t ob = ((size_t)bidx * NQ + qrow) * DIMM + h * HD;
        attn_out[ob + ln]      = (__bf16)v0;
        attn_out[ob + 16 + ln] = (__bf16)v1;
    }
}

// ---------- output projection: [12544,192] x [192,192] + bias ----------
__global__ __launch_bounds__(256)
void proj_gemm(const __bf16* __restrict__ attn, const __bf16* __restrict__ wt,
               const float* __restrict__ bias, float* __restrict__ out) {
    const int lane = threadIdx.x & 31;
    const int ln   = lane & 15;
    const int half = lane >> 4;
    const int gw   = (blockIdx.x * blockDim.x + threadIdx.x) >> 5;
    const int mt   = gw / 6;   // 784 m-tiles
    const int nt   = gw % 6;   // 6 n-tiles of 32
    if (mt >= 784) return;
    const int row = mt * 16 + ln;
    const int n0  = nt * 32;
    const int ca  = half ? 16 : 0;
    const int cb  = half ? 16 : 0;
    v8f acc0 = {}, acc1 = {};
#pragma unroll
    for (int kk = 0; kk < DIMM; kk += 32) {
        const __bf16* ap = attn + (size_t)row * DIMM + kk + ca;
        v8bf alo = *(const v8bf*)ap;
        v8bf ahi = *(const v8bf*)(ap + 8);
        v16bf a;
#pragma unroll
        for (int i = 0; i < 8; i++) { a[i] = alo[i]; a[8 + i] = ahi[i]; }
        v16bf b0 = *(const v16bf*)(wt + (size_t)(n0 + ln) * DIMM + kk + cb);
        v16bf b1 = *(const v16bf*)(wt + (size_t)(n0 + 16 + ln) * DIMM + kk + cb);
        acc0 = __builtin_amdgcn_wmma_f32_16x16x32_bf16(false, a, false, b0, (short)0, acc0, false, false);
        acc1 = __builtin_amdgcn_wmma_f32_16x16x32_bf16(false, a, false, b1, (short)0, acc1, false, false);
    }
#pragma unroll
    for (int sub = 0; sub < 2; sub++) {
        v8f& acc = sub ? acc1 : acc0;
        int col = n0 + sub * 16 + ln;
        float bv = bias[col];
#pragma unroll
        for (int r = 0; r < 8; r++) {
            int m = mt * 16 + r + half * 8;
            out[(size_t)m * DIMM + col] = acc[r] + bv;
        }
    }
}

extern "C" void kernel_launch(void* const* d_in, const int* in_sizes, int n_in,
                              void* d_out, int out_size, void* d_ws, size_t ws_size,
                              hipStream_t stream) {
    const float* x      = (const float*)d_in[0];
    const float* w_qkv  = (const float*)d_in[3];
    const float* b_qkv  = (const float*)d_in[4];
    const float* wp_q   = (const float*)d_in[5];
    const float* g_q    = (const float*)d_in[6];
    const float* bn_q   = (const float*)d_in[7];
    const float* wp_k   = (const float*)d_in[8];
    const float* g_k    = (const float*)d_in[9];
    const float* bn_k   = (const float*)d_in[10];
    const float* wp_v   = (const float*)d_in[11];
    const float* g_v    = (const float*)d_in[12];
    const float* bn_v   = (const float*)d_in[13];
    const float* w_proj = (const float*)d_in[14];
    const float* b_proj = (const float*)d_in[15];
    float* out = (float*)d_out;

    char* ws = (char*)d_ws;
    size_t off = 0;
    auto alloc = [&](size_t bytes) -> char* {
        char* p = ws + off;
        off = (off + bytes + 255) & ~(size_t)255;
        return p;
    };
    float*  qkv_f   = (float*)alloc((size_t)3 * BH * NQ * HD * 4);   // [3][B][h][NQ][32]
    __bf16* q_bf    = (__bf16*)alloc((size_t)BH * NQ * HD * 2);
    float*  q_f     = (float*)alloc((size_t)BH * NQ * HD * 4);       // residual
    __bf16* k_bf    = (__bf16*)alloc((size_t)BH * NK * HD * 2);
    __bf16* v_t     = (__bf16*)alloc((size_t)BH * NK * HD * 2);      // transposed
    __bf16* attn_bf = (__bf16*)alloc((size_t)BB * NQ * DIMM * 2);
    __bf16* wqkv_t  = (__bf16*)alloc((size_t)576 * DIMM * 2);
    __bf16* wproj_t = (__bf16*)alloc((size_t)DIMM * DIMM * 2);

    prep_weights<<<(576 * DIMM + DIMM * DIMM + 255) / 256, 256, 0, stream>>>(w_qkv, w_proj, wqkv_t, wproj_t);
    qkv_gemm<<<(784 * 18 * 32) / 256, 256, 0, stream>>>(x, wqkv_t, b_qkv, qkv_f);

    const float* qsrc = qkv_f;
    const float* ksrc = qkv_f + (size_t)BH * NQ * HD;
    const float* vsrc = qkv_f + (size_t)2 * BH * NQ * HD;
    pool_ln<<<(BH * NQ * 32 + 255) / 256, 256, 0, stream>>>(qsrc, wp_q, g_q, bn_q, 1, HH, WW, q_bf, 0, q_f);
    pool_ln<<<(BH * NK * 32 + 255) / 256, 256, 0, stream>>>(ksrc, wp_k, g_k, bn_k, 2, HPK, HPK, k_bf, 0, nullptr);
    pool_ln<<<(BH * NK * 32 + 255) / 256, 256, 0, stream>>>(vsrc, wp_v, g_v, bn_v, 2, HPK, HPK, v_t, 1, nullptr);

    attention<<<(BH * 196) / 8, 256, 0, stream>>>(q_bf, q_f, k_bf, v_t, attn_bf);
    proj_gemm<<<(784 * 6 * 32) / 256, 256, 0, stream>>>(attn_bf, wproj_t, b_proj, out);
}
